// LeadingZeroDetector57_76312978916100
// MI455X (gfx1250) — 8-bit matrix operand order repaired and checked by reading, compiled, not run on hardware
//
#include <hip/hip_runtime.h>

typedef int v8i __attribute__((ext_vector_type(8)));
typedef unsigned int uint32;

#define NB 57
#define ROWS_PER_BLOCK 256

__global__ __launch_bounds__(256)
void lzc57_kernel(const float* __restrict__ X, float* __restrict__ out, int nrows) {
    __shared__ __align__(16) uint32 tile[ROWS_PER_BLOCK * NB];

    const int tid  = threadIdx.x;
    const int lane = tid & 31;
    const int base = blockIdx.x * ROWS_PER_BLOCK;
    const int rows = min(ROWS_PER_BLOCK, nrows - base);

    {
        const unsigned long long gaddr =
            (unsigned long long)X + (unsigned long long)base * (NB * 4ull);
        const uint32 ldsbase = (uint32)(unsigned long long)(uintptr_t)&tile[0];
        const int dwords = rows * NB;
        const int chunks = dwords >> 2;
        for (int c = tid; c < chunks; c += ROWS_PER_BLOCK) {
            uint32 goff = (uint32)(c << 4);
            uint32 loff = ldsbase + goff;
            asm volatile("global_load_async_to_lds_b128 %0, %1, %2"
                         :: "v"(loff), "v"(goff), "s"(gaddr) : "memory");
        }
        const int rem = dwords & 3;
        if (tid < rem) {
            uint32 goff = (uint32)((chunks << 4) + (tid << 2));
            uint32 loff = ldsbase + goff;
            asm volatile("global_load_async_to_lds_b32 %0, %1, %2"
                         :: "v"(loff), "v"(goff), "s"(gaddr) : "memory");
        }
        asm volatile("s_wait_asynccnt 0x0" ::: "memory");
    }
    __syncthreads();

    uint32 lz = 57u;
    if (tid < rows) {
        const uint32* rp = &tile[tid * NB];
        uint32 hi = 0u, lo = 0u;
#pragma unroll
        for (int i = 0; i < 32; ++i)
            hi |= ((rp[i] >> 23) & 1u) << (31 - i);
#pragma unroll
        for (int t = 0; t < 25; ++t)
            lo |= ((rp[32 + t] >> 23) & 1u) << (24 - t);
        uint32 v = lo << 7;
        lz = hi ? (uint32)__clz((int)hi)
                : (v ? (uint32)(32 + __clz((int)v)) : 57u);
    }

    const int wrow = base + (tid & ~31);
    const bool waveFull = (wrow + 32) <= (base + rows);

    if (waveFull) {
        const int h   = lane >> 4;
        const int col = lane & 15;

        v8i bmat;
#pragma unroll
        for (int vg = 0; vg < 8; ++vg) {
            uint32 w = 0u;
#pragma unroll
            for (int t = 0; t < 4; ++t) {
                int k = ((vg >> 2) << 5) + (h << 4) + ((vg & 3) << 2) + t;
                uint32 bit = (col < 6) ? (uint32)((k >> (5 - col)) & 1) : 0u;
                w |= bit << (t * 8);
            }
            bmat[vg] = (int)w;
        }

        int lz0 = __shfl((int)lz, col, 32);
        int lz1 = __shfl((int)lz, col + 16, 32);
        v8i a0, a1;
#pragma unroll
        for (int vg = 0; vg < 8; ++vg) {
            int ks = ((vg >> 2) << 5) + (h << 4) + ((vg & 3) << 2);
            int d0 = lz0 - ks;
            int d1 = lz1 - ks;
            a0[vg] = (d0 >= 0 && d0 < 4) ? (int)(1u << (d0 * 8)) : 0;
            a1[vg] = (d1 >= 0 && d1 < 4) ? (int)(1u << (d1 * 8)) : 0;
        }

        v8i zero = {};
        v8i d0 = __builtin_amdgcn_wmma_i32_16x16x64_iu8(false, a0, false, bmat, zero, false, false);
        v8i d1 = __builtin_amdgcn_wmma_i32_16x16x64_iu8(false, a1, false, bmat, zero, false, false);

        if (col < 6) {
#pragma unroll
            for (int g = 0; g < 8; ++g) {
                out[(size_t)(wrow +      g + 8 * h) * 6 + col] = (float)d0[g];
                out[(size_t)(wrow + 16 + g + 8 * h) * 6 + col] = (float)d1[g];
            }
        }
    } else if (tid < rows) {
        size_t o = (size_t)(base + tid) * 6;
#pragma unroll
        for (int j = 0; j < 6; ++j)
            out[o + j] = (float)((lz >> (5 - j)) & 1u);
    }
}

extern "C" void kernel_launch(void* const* d_in, const int* in_sizes, int n_in,
                              void* d_out, int out_size, void* d_ws, size_t ws_size,
                              hipStream_t stream) {
    (void)n_in; (void)out_size; (void)d_ws; (void)ws_size;
    const float* X = (const float*)d_in[0];
    float* out = (float*)d_out;
    int nrows = in_sizes[0] / NB;
    int blocks = (nrows + ROWS_PER_BLOCK - 1) / ROWS_PER_BLOCK;
    lzc57_kernel<<<blocks, ROWS_PER_BLOCK, 0, stream>>>(X, out, nrows);
}
